// GIN_52639119180538
// MI455X (gfx1250) — matrix operand order repaired and checked by reading, not compiled, not run on hardware
//
#include <hip/hip_runtime.h>
#include <hip/hip_bf16.h>
#include <stdint.h>

// ---------------------------------------------------------------------------
// Types for CDNA5 WMMA (wave32): A/B = v16h (16 halves / lane), C/D = v8f.
// ---------------------------------------------------------------------------
typedef __attribute__((ext_vector_type(16))) _Float16 v16h;
typedef __attribute__((ext_vector_type(8)))  float    v8f;

struct Frag32B { uint4 lo, hi; };

static __device__ __forceinline__ v16h load_frag16(const _Float16* p0, const _Float16* p1) {
  Frag32B t;
  t.lo = *(const uint4*)p0;   // 8 halves
  t.hi = *(const uint4*)p1;   // 8 halves
  return __builtin_bit_cast(v16h, t);
}

// Exact pointee type the async-to-LDS builtin expects (per hipcc diagnostic):
// pointer to __attribute__((vector_size(16))) int, in AS(1)/AS(3).
typedef int vi4 __attribute__((__vector_size__(4 * sizeof(int))));
typedef __attribute__((address_space(1))) vi4 gvi4;
typedef __attribute__((address_space(3))) vi4 lvi4;

// Async global -> LDS copy of 16 bytes (tracked by ASYNCcnt, not LOADcnt).
static __device__ __forceinline__ void stage_async16(const _Float16* g, _Float16* l) {
#if __has_builtin(__builtin_amdgcn_global_load_async_to_lds_b128)
  __builtin_amdgcn_global_load_async_to_lds_b128(
      (gvi4*)(uintptr_t)g,   // flat value == global value -> inttoptr to AS(1)
      (lvi4*)l,              // C-style cast: addrspacecast generic -> AS(3)
      0, 0);
#else
  unsigned int loff =
      (unsigned int)(uintptr_t)(__attribute__((address_space(3))) _Float16*)l;
  asm volatile("global_load_async_to_lds_b128 %0, %1, off"
               :: "v"(loff), "v"((unsigned long long)(uintptr_t)g) : "memory");
#endif
}

static __device__ __forceinline__ void wait_async0() {
  asm volatile("s_wait_asynccnt 0x0" ::: "memory");
}

// ---------------------------------------------------------------------------
// agg[i] = (1 + eps) * x[i]
// ---------------------------------------------------------------------------
__global__ __launch_bounds__(256) void k_init_agg(const float* __restrict__ x,
                                                  const float* __restrict__ eps,
                                                  float* __restrict__ agg, int total) {
  int i = blockIdx.x * blockDim.x + threadIdx.x;
  if (i < total) agg[i] = (1.0f + eps[0]) * x[i];
}

// ---------------------------------------------------------------------------
// segment_sum scatter: agg[dst, f] += x[src, f]   (one thread per (edge, f))
// ---------------------------------------------------------------------------
__global__ __launch_bounds__(256) void k_edge_scatter(const float* __restrict__ x,
                                                      const int* __restrict__ ei,
                                                      float* __restrict__ agg,
                                                      int E, int F, int fshift) {
  int idx = blockIdx.x * blockDim.x + threadIdx.x;
  if (idx >= E * F) return;
  int e = idx >> fshift;
  int f = idx & (F - 1);
  int s = ei[e];        // edge_index[0][e]
  int d = ei[E + e];    // edge_index[1][e]
  atomicAdd(&agg[(size_t)d * F + f], x[(size_t)s * F + f]);
}

// ---------------------------------------------------------------------------
// f32 -> f16 convert, zero-fill padded rows (rows >= Mrows).
// ---------------------------------------------------------------------------
__global__ __launch_bounds__(256) void k_cvt_f16(const float* __restrict__ src,
                                                 _Float16* __restrict__ dst,
                                                 int Mrows, int total, int cshift) {
  int i = blockIdx.x * blockDim.x + threadIdx.x;
  if (i >= total) return;
  int r = i >> cshift;
  dst[i] = (r < Mrows) ? (_Float16)src[i] : (_Float16)0.0f;
}

// ---------------------------------------------------------------------------
// WMMA GEMM:  C[M,N] = epilogue(A[Mpad,K] @ B[K,N] + bias)
//   Block = 128 threads = 4 waves; block tile 128(M) x 64(N).
//   Each wave computes TWO 16x64 m-tiles (rows base and base+64), so every
//   B fragment read from LDS feeds two independent WMMAs.
//   B tile (32x64 f16 per K-step) staged to LDS via async-to-LDS copies,
//   double buffered; A fragments prefetched into registers.
//   Epilogue: ReLU, then optional BatchNorm affine (fuse_bn).
// Requirements: K % 32 == 0, N % 64 == 0, Mpad % 128 == 0.
// ---------------------------------------------------------------------------
#define BSTRIDE 72   // 64 + 8 halves of padding per LDS row

__global__ __launch_bounds__(128) void k_gemm_wmma(
    const _Float16* __restrict__ A, const _Float16* __restrict__ B,
    const float* __restrict__ bias, float* __restrict__ C,
    int M, int K, int N, int ngrp, int fuse_bn,
    const float* __restrict__ g, const float* __restrict__ be,
    const float* __restrict__ rm, const float* __restrict__ rv) {
  __shared__ _Float16 Bs[2][32 * BSTRIDE];

  const int lane = threadIdx.x & 31;
  const int wave = threadIdx.x >> 5;               // 0..3
  const int bm = blockIdx.x / ngrp;
  const int tn = (blockIdx.x - bm * ngrp) << 6;    // starting column of block

  // Staging map: thread t copies 16 halves (2 x 16B): row = t>>2, col = (t&3)*16.
  const int sr = threadIdx.x >> 2;                 // 0..31
  const int sc = (threadIdx.x & 3) << 4;           // 0,16,32,48
  const _Float16* sg = B + (size_t)sr * N + tn + sc;   // advances 32*N per stage
  _Float16* sl0 = &Bs[0][sr * BSTRIDE + sc];
  _Float16* sl1 = &Bs[1][sr * BSTRIDE + sc];

  // A-fragment addressing per ISA f16 16x32 layout:
  //   lane&15 = M row, (lane>>4)*8 = K sub-offset; halves [0..7]=K+0..7, [8..15]=K+16..23
  const int arow0 = (bm << 7) + (wave << 4) + (lane & 15);
  const int arow1 = arow0 + 64;
  const int koff = (lane >> 4) << 3;
  const _Float16* ap0 = A + (size_t)arow0 * K + koff;
  const _Float16* ap1 = A + (size_t)arow1 * K + koff;

  v8f acc0[4], acc1[4];
#pragma unroll
  for (int j = 0; j < 4; ++j) { acc0[j] = v8f{}; acc1[j] = v8f{}; }

  const int nk = K >> 5;

  // Prologue: stage K-step 0 into buffer 0; load first A fragments.
  stage_async16(sg, sl0);
  stage_async16(sg + 8, sl0 + 8);
  const _Float16* bsrc = sg + (size_t)32 * N;
  v16h aCur0 = load_frag16(ap0, ap0 + 16);
  v16h aCur1 = load_frag16(ap1, ap1 + 16);
  v16h aNext0 = aCur0, aNext1 = aCur1;

  for (int i = 0; i < nk; ++i) {
    const int cur = i & 1;
    // Wait for this wave's async stage of buffer `cur`, then make it visible.
    wait_async0();
    __syncthreads();
    // Issue next stage + next A fragments while we compute on the current step.
    if (i + 1 < nk) {
      _Float16* dst = cur ? sl0 : sl1;
      stage_async16(bsrc, dst);
      stage_async16(bsrc + 8, dst + 8);
      bsrc += (size_t)32 * N;
      aNext0 = load_frag16(ap0 + 32, ap0 + 48);
      aNext1 = load_frag16(ap1 + 32, ap1 + 48);
      ap0 += 32;
      ap1 += 32;
    }
    // B fragments from LDS: lane&15 = N col; halves [0..7]=K 8h+0..7, [8..15]=K 16+8h+0..7.
    const _Float16* rB = &Bs[cur][((lane >> 4) << 3) * BSTRIDE + (lane & 15)];
    v16h b0, b1, b2, b3;
#pragma unroll
    for (int s = 0; s < 16; ++s) {
      const _Float16* p = rB + ((s < 8) ? s : (s + 8)) * BSTRIDE;
      b0[s] = p[0];
      b1[s] = p[16];
      b2[s] = p[32];
      b3[s] = p[48];
    }
    acc0[0] = __builtin_amdgcn_wmma_f32_16x16x32_f16(false, aCur0, false, b0, (short)0, acc0[0], false, false);
    acc1[0] = __builtin_amdgcn_wmma_f32_16x16x32_f16(false, aCur1, false, b0, (short)0, acc1[0], false, false);
    acc0[1] = __builtin_amdgcn_wmma_f32_16x16x32_f16(false, aCur0, false, b1, (short)0, acc0[1], false, false);
    acc1[1] = __builtin_amdgcn_wmma_f32_16x16x32_f16(false, aCur1, false, b1, (short)0, acc1[1], false, false);
    acc0[2] = __builtin_amdgcn_wmma_f32_16x16x32_f16(false, aCur0, false, b2, (short)0, acc0[2], false, false);
    acc1[2] = __builtin_amdgcn_wmma_f32_16x16x32_f16(false, aCur1, false, b2, (short)0, acc1[2], false, false);
    acc0[3] = __builtin_amdgcn_wmma_f32_16x16x32_f16(false, aCur0, false, b3, (short)0, acc0[3], false, false);
    acc1[3] = __builtin_amdgcn_wmma_f32_16x16x32_f16(false, aCur1, false, b3, (short)0, acc1[3], false, false);
    aCur0 = aNext0;
    aCur1 = aNext1;
  }

  // C layout: col = lane&15 (+16j), row = rbase + r (r = accumulator VGPR idx)
  const int rbase0 = (bm << 7) + (wave << 4) + ((lane >> 4) << 3);
  const int rbase1 = rbase0 + 64;
#pragma unroll
  for (int j = 0; j < 4; ++j) {
    const int col = tn + (j << 4) + (lane & 15);
    const float bv = bias[col];
    float sc2 = 1.0f, mu = 0.0f, bb = 0.0f;
    if (fuse_bn) {
      sc2 = g[col] * rsqrtf(rv[col] + 1e-5f);
      mu = rm[col];
      bb = be[col];
    }
#pragma unroll
    for (int r = 0; r < 8; ++r) {
      const int m0 = rbase0 + r;
      if (m0 < M) {
        float v = fmaxf(acc0[j][r] + bv, 0.0f);
        if (fuse_bn) v = (v - mu) * sc2 + bb;
        C[(size_t)m0 * N + col] = v;
      }
      const int m1 = rbase1 + r;
      if (m1 < M) {
        float v = fmaxf(acc1[j][r] + bv, 0.0f);
        if (fuse_bn) v = (v - mu) * sc2 + bb;
        C[(size_t)m1 * N + col] = v;
      }
    }
  }
}

// ---------------------------------------------------------------------------
// Final head: out[t, :] = (h[a] * h[b]) @ Wf + bf   for train edges.
// ---------------------------------------------------------------------------
__global__ __launch_bounds__(256) void k_edge_out(const float* __restrict__ h,
                                                  const int* __restrict__ ei,
                                                  const int* __restrict__ teid,
                                                  const float* __restrict__ Wf,
                                                  const float* __restrict__ bf,
                                                  float* __restrict__ out,
                                                  int E, int Hdim, int C, int T) {
  int t = blockIdx.x * blockDim.x + threadIdx.x;
  if (t >= T) return;
  int e = teid[t];
  int a = ei[e];
  int b = ei[E + e];
  const float* ha = h + (size_t)a * Hdim;
  const float* hb = h + (size_t)b * Hdim;
  float acc[8];
#pragma unroll
  for (int c = 0; c < 8; ++c) acc[c] = (c < C) ? bf[c] : 0.0f;
  for (int k = 0; k < Hdim; ++k) {
    const float p = ha[k] * hb[k];
    const float* wr = Wf + (size_t)k * C;
#pragma unroll
    for (int c = 0; c < 8; ++c)
      if (c < C) acc[c] = fmaf(p, wr[c], acc[c]);
  }
  float* o = out + (size_t)t * C;
#pragma unroll
  for (int c = 0; c < 8; ++c)
    if (c < C) o[c] = acc[c];
}

// ---------------------------------------------------------------------------
// Host-side orchestration
// ---------------------------------------------------------------------------
static inline int ilog2i(int v) { int s = 0; while ((1 << s) < v) ++s; return s; }

extern "C" void kernel_launch(void* const* d_in, const int* in_sizes, int n_in,
                              void* d_out, int out_size, void* d_ws, size_t ws_size,
                              hipStream_t stream) {
  const float* x    = (const float*)d_in[0];
  const int*   ei   = (const int*)  d_in[1];
  const int*   teid = (const int*)  d_in[2];
  const float* eps1 = (const float*)d_in[3];
  const float* W1a  = (const float*)d_in[4];
  const float* b1a  = (const float*)d_in[5];
  const float* W1b  = (const float*)d_in[6];
  const float* b1b  = (const float*)d_in[7];
  const float* g1   = (const float*)d_in[8];
  const float* be1  = (const float*)d_in[9];
  const float* rm1  = (const float*)d_in[10];
  const float* rv1  = (const float*)d_in[11];
  const float* eps2 = (const float*)d_in[12];
  const float* W2a  = (const float*)d_in[13];
  const float* b2a  = (const float*)d_in[14];
  const float* g2   = (const float*)d_in[15];
  const float* be2  = (const float*)d_in[16];
  const float* rm2  = (const float*)d_in[17];
  const float* rv2  = (const float*)d_in[18];
  const float* Wl   = (const float*)d_in[19];
  const float* bl   = (const float*)d_in[20];
  const float* Wf   = (const float*)d_in[21];
  const float* bf   = (const float*)d_in[22];
  float* out = (float*)d_out;

  const int H = in_sizes[5];             // 512
  const int F = in_sizes[4] / H;         // 128
  const int N = in_sizes[0] / F;         // 10000
  const int E = in_sizes[1] / 2;         // 160000
  const int T = in_sizes[2];             // 40000
  const int C = in_sizes[22];            // 7
  const int Mpad = (N + 127) & ~127;     // block tile is 128 rows
  const int fs = ilog2i(F), hs = ilog2i(H);

  // Workspace carve-out
  char* ws = (char*)d_ws;
  size_t off = 0;
  auto carve = [&](size_t bytes) -> void* {
    off = (off + 255) & ~(size_t)255;
    void* p = ws + off;
    off += bytes;
    return p;
  };
  _Float16* w1a16 = (_Float16*)carve((size_t)F * H * 2);
  _Float16* w1b16 = (_Float16*)carve((size_t)H * H * 2);
  _Float16* w2a16 = (_Float16*)carve((size_t)H * H * 2);
  _Float16* wl16  = (_Float16*)carve((size_t)H * H * 2);
  _Float16* a16   = (_Float16*)carve((size_t)Mpad * H * 2);
  float*    bufA  = (float*)   carve((size_t)Mpad * H * 4);
  float*    bufB  = (float*)   carve((size_t)Mpad * H * 4);
  (void)ws_size;

  const int TB = 256;
  const int TBG = 128;                           // GEMM: 4 waves/block
  auto blocks = [](long long n, int tb) { return (int)((n + tb - 1) / tb); };
  const int ngrp = H >> 6;                       // 64-col groups
  const int gemm_blks = (Mpad >> 7) * ngrp;      // one block per 128x64 tile

  // Weight conversion fp32 -> f16
  k_cvt_f16<<<blocks((long long)F * H, TB), TB, 0, stream>>>(W1a, w1a16, F, F * H, hs);
  k_cvt_f16<<<blocks((long long)H * H, TB), TB, 0, stream>>>(W1b, w1b16, H, H * H, hs);
  k_cvt_f16<<<blocks((long long)H * H, TB), TB, 0, stream>>>(W2a, w2a16, H, H * H, hs);
  k_cvt_f16<<<blocks((long long)H * H, TB), TB, 0, stream>>>(Wl,  wl16,  H, H * H, hs);

  // --- Layer 1: GIN aggregation on x (F cols) ---
  k_init_agg<<<blocks((long long)N * F, TB), TB, 0, stream>>>(x, eps1, bufA, N * F);
  k_edge_scatter<<<blocks((long long)E * F, TB), TB, 0, stream>>>(x, ei, bufA, E, F, fs);
  k_cvt_f16<<<blocks((long long)Mpad * F, TB), TB, 0, stream>>>(bufA, a16, N, Mpad * F, fs);

  // GEMM1: [Mpad,F] @ [F,H], ReLU                            -> bufB
  k_gemm_wmma<<<gemm_blks, TBG, 0, stream>>>(a16, w1a16, b1a, bufB, N, F, H, ngrp, 0,
                                             nullptr, nullptr, nullptr, nullptr);
  k_cvt_f16<<<blocks((long long)Mpad * H, TB), TB, 0, stream>>>(bufB, a16, N, Mpad * H, hs);

  // GEMM2: [Mpad,H] @ [H,H], ReLU + BN1                      -> bufA
  k_gemm_wmma<<<gemm_blks, TBG, 0, stream>>>(a16, w1b16, b1b, bufA, N, H, H, ngrp, 1,
                                             g1, be1, rm1, rv1);

  // --- Layer 2: GIN aggregation on bufA (H cols) ---
  k_init_agg<<<blocks((long long)N * H, TB), TB, 0, stream>>>(bufA, eps2, bufB, N * H);
  k_edge_scatter<<<blocks((long long)E * H, TB), TB, 0, stream>>>(bufA, ei, bufB, E, H, hs);
  k_cvt_f16<<<blocks((long long)Mpad * H, TB), TB, 0, stream>>>(bufB, a16, N, Mpad * H, hs);

  // GEMM3: ReLU + BN2                                        -> bufA
  k_gemm_wmma<<<gemm_blks, TBG, 0, stream>>>(a16, w2a16, b2a, bufA, N, H, H, ngrp, 1,
                                             g2, be2, rm2, rv2);
  k_cvt_f16<<<blocks((long long)Mpad * H, TB), TB, 0, stream>>>(bufA, a16, N, Mpad * H, hs);

  // GEMM4: ReLU                                              -> bufB (final embeddings)
  k_gemm_wmma<<<gemm_blks, TBG, 0, stream>>>(a16, wl16, bl, bufB, N, H, H, ngrp, 0,
                                             nullptr, nullptr, nullptr, nullptr);

  // Edge head
  k_edge_out<<<blocks(T, TB), TB, 0, stream>>>(bufB, ei, teid, Wf, bf, out, E, H, C, T);
  (void)out_size; (void)n_in;
}
